// SamplingBottleneckModule_61710090109684
// MI455X (gfx1250) — matrix operand order repaired and checked by reading, not compiled, not run on hardware
//
#include <hip/hip_runtime.h>

// Fused SamplingBottleneck forward for MI455X (gfx1250, wave32, WMMA).
//
// Roofline: ~258 MB HBM traffic (x in, 3 outputs out) -> ~11 us floor @23.3TB/s.
// 34.4 GFLOP of GEMM via v_wmma_f32_16x16x32_f16 is below that floor, so the
// design keeps logits LDS-resident (one fused pass), reads x once, writes each
// output once. Weights (1 MB as f16) stay hot in the 192 MB L2.

typedef __attribute__((ext_vector_type(16))) _Float16 v16h;
typedef __attribute__((ext_vector_type(8)))  _Float16 v8h;
typedef __attribute__((ext_vector_type(8)))  float    v8f;

namespace {

constexpr int kF        = 512;          // feature dim (GEMM K)
constexpr int kN        = 512;          // class dim (GEMM N)
constexpr int kRows     = 16;           // rows per workgroup (GEMM M tile)
constexpr int kBT       = 16 * 2048;    // B*T rows
constexpr int kLXStride = 520;          // f16 x-tile stride: 260 dwords/row -> bank-conflict-free A loads

union AFrag { v16h v; v8h h[2]; };
union BFrag { v16h v; v8h h[2]; _Float16 e[16]; };

__global__ __launch_bounds__(256) void convert_w_kernel(
    const float* __restrict__ wb, const float* __restrict__ wv,
    _Float16* __restrict__ o) {
  int i = blockIdx.x * 256 + threadIdx.x;
  if (i < kN * kF) {
    o[i]           = (_Float16)wb[i];
    o[kN * kF + i] = (_Float16)wv[i];
  }
}

template <bool USE_F16W>
__global__ __launch_bounds__(256) void fused_kernel(
    const float* __restrict__ x,
    const _Float16* __restrict__ w16,   // [w_both | w_values] as f16 (USE_F16W path)
    const float* __restrict__ w32b,     // fp32 fallback (ws too small)
    const float* __restrict__ w32v,
    float* __restrict__ out_marg,
    float* __restrict__ out_val,
    float* __restrict__ out_cum) {
  // 64 KB LDS: lb = logits_both -> probs; lv = logits_values -> cumsum stage.
  // x f16 tile overlaps lv (x fully consumed before lv is first written).
  __shared__ __align__(16) unsigned char smem[65536];
  float*    lb = (float*)smem;                 // 16 x 512 f32
  float*    lv = (float*)(smem + 32768);       // 16 x 512 f32
  _Float16* lx = (_Float16*)(smem + 32768);    // 16 x 520 f16 (16.3 KB, overlaps lv)

  const int tid  = threadIdx.x;
  const int row0 = blockIdx.x * kRows;

  // ---- stage x tile (fp32 -> f16), coalesced ----
  for (int i = tid; i < kRows * kF; i += 256) {
    int r = i >> 9, f = i & (kF - 1);
    lx[r * kLXStride + f] = (_Float16)x[(size_t)(row0 + r) * kF + f];
  }
  __syncthreads();

  const int wid  = tid >> 5;    // wave 0..7: owns N columns [wid*64, wid*64+64)
  const int lane = tid & 31;
  const int l16  = lane & 15;
  const int half = lane >> 4;

  // ---- GEMM: 8 output tiles/wave (4 N-tiles x 2 weight matrices) ----
  v8f zero = {};
  v8f accB[4], accV[4];
#pragma unroll
  for (int t = 0; t < 4; ++t) { accB[t] = zero; accV[t] = zero; }

  // A-frag lane addressing: row = l16, K = half*16 + {0..15} (matches B-frag)
  const _Float16* axp = lx + l16 * kLXStride + half * 16;

  for (int k0 = 0; k0 < kF; k0 += 32) {
    AFrag a;
    a.h[0] = *(const v8h*)(axp + k0);
    a.h[1] = *(const v8h*)(axp + k0 + 8);
#pragma unroll
    for (int t = 0; t < 4; ++t) {
      // B-frag: lane holds column n = wid*64+t*16+l16, K = k0 + half*16 + {0..15}
      size_t noff = (size_t)(wid * 64 + t * 16 + l16) * kF + (size_t)(k0 + half * 16);
      BFrag b;
      if (USE_F16W) {
        b.h[0] = *(const v8h*)(w16 + noff);
        b.h[1] = *(const v8h*)(w16 + noff + 8);
      } else {
#pragma unroll
        for (int j = 0; j < 16; ++j) b.e[j] = (_Float16)w32b[noff + j];
      }
      accB[t] = __builtin_amdgcn_wmma_f32_16x16x32_f16(
          false, a.v, false, b.v, (short)0, accB[t], false, false);
      if (USE_F16W) {
        b.h[0] = *(const v8h*)(w16 + (size_t)kN * kF + noff);
        b.h[1] = *(const v8h*)(w16 + (size_t)kN * kF + noff + 8);
      } else {
#pragma unroll
        for (int j = 0; j < 16; ++j) b.e[j] = (_Float16)w32v[noff + j];
      }
      accV[t] = __builtin_amdgcn_wmma_f32_16x16x32_f16(
          false, a.v, false, b.v, (short)0, accV[t], false, false);
    }
  }
  __syncthreads();   // all lx reads done before lv (aliased) is written

  // ---- spill accumulators to LDS (D layout: VGPR r -> M = r + half*8, N = l16) ----
#pragma unroll
  for (int t = 0; t < 4; ++t) {
    int ncol = wid * 64 + t * 16 + l16;
#pragma unroll
    for (int r = 0; r < 8; ++r) {
      int m = r + half * 8;
      lb[m * kN + ncol] = accB[t][r];
      lv[m * kN + ncol] = accV[t][r];
    }
  }
  __syncthreads();

  // ---- epilogue: wave w -> rows 2w (lanes 0-15) and 2w+1 (lanes 16-31) ----
  const int row = 2 * wid + half;
  float* lbrow = lb + row * kN;
  float* lvrow = lv + row * kN;
  const size_t orow = (size_t)(row0 + row) * kN;

  float t32[32], pv[32], yv[32];

  // softmax(logits_both) -> probs (kept in lb), y = log(1-p) in regs
  float mx = -3.0e38f;
#pragma unroll
  for (int j = 0; j < 32; ++j) { t32[j] = lbrow[l16 + 16 * j]; mx = fmaxf(mx, t32[j]); }
#pragma unroll
  for (int off = 1; off < 16; off <<= 1) mx = fmaxf(mx, __shfl_xor(mx, off, 32));
  float s = 0.f;
#pragma unroll
  for (int j = 0; j < 32; ++j) { t32[j] = __expf(t32[j] - mx); s += t32[j]; }
#pragma unroll
  for (int off = 1; off < 16; off <<= 1) s += __shfl_xor(s, off, 32);
  float inv = 1.0f / s;
#pragma unroll
  for (int j = 0; j < 32; ++j) {
    float p = t32[j] * inv;
    pv[j] = p;
    lbrow[l16 + 16 * j] = p;       // own-lane slots only: no barrier needed yet
    yv[j] = __logf(1.0f - p);
  }

  // values = softmax(probs + logits_values)
  mx = -3.0e38f;
#pragma unroll
  for (int j = 0; j < 32; ++j) { t32[j] = pv[j] + lvrow[l16 + 16 * j]; mx = fmaxf(mx, t32[j]); }
#pragma unroll
  for (int off = 1; off < 16; off <<= 1) mx = fmaxf(mx, __shfl_xor(mx, off, 32));
  s = 0.f;
#pragma unroll
  for (int j = 0; j < 32; ++j) { t32[j] = __expf(t32[j] - mx); s += t32[j]; }
#pragma unroll
  for (int off = 1; off < 16; off <<= 1) s += __shfl_xor(s, off, 32);
  inv = 1.0f / s;
#pragma unroll
  for (int j = 0; j < 32; ++j) out_val[orow + l16 + 16 * j] = t32[j] * inv;

  // 3-step Newton for alpha: err = sum(exp(a*y)) + (K - N); d = sum(exp(a*y)*y)
  float alpha = 8.0f;
#pragma unroll
  for (int it = 0; it < 3; ++it) {
    float s1 = 0.f, s2 = 0.f;
#pragma unroll
    for (int j = 0; j < 32; ++j) { float e = __expf(alpha * yv[j]); s1 += e; s2 += e * yv[j]; }
#pragma unroll
    for (int off = 1; off < 16; off <<= 1) {
      s1 += __shfl_xor(s1, off, 32);
      s2 += __shfl_xor(s2, off, 32);
    }
    alpha -= (s1 + (8.0f - 512.0f)) / s2;
  }
  // marginals = 1 - (1-p)^alpha = 1 - exp(alpha*y)
#pragma unroll
  for (int j = 0; j < 32; ++j) out_marg[orow + l16 + 16 * j] = 1.0f - __expf(alpha * yv[j]);

  __syncthreads();   // cumsum reads other lanes' probs; also frees lv for staging

  // exclusive cumsum of probs: lane owns contiguous 32-chunk, scan lane totals
  {
    float c32[32]; float run = 0.f;
    const float* pr = lb + row * kN + l16 * 32;
#pragma unroll
    for (int j = 0; j < 32; ++j) { c32[j] = run; run += pr[j]; }
    float tot = run, scan = run;
#pragma unroll
    for (int off = 1; off < 16; off <<= 1) {
      float nb = __shfl_up(scan, off, 16);
      if (l16 >= off) scan += nb;
    }
    float base = scan - tot;   // exclusive prefix of lane totals
    float* co = lv + row * kN + l16 * 32;
#pragma unroll
    for (int j = 0; j < 32; ++j) co[j] = base + c32[j];
  }
  __syncthreads();
  // coalesced stream-out of cumsum
  for (int i = tid; i < kRows * kN; i += 256)
    out_cum[(size_t)row0 * kN + i] = lv[i];
}

}  // namespace

extern "C" void kernel_launch(void* const* d_in, const int* in_sizes, int n_in,
                              void* d_out, int out_size, void* d_ws, size_t ws_size,
                              hipStream_t stream) {
  (void)in_sizes; (void)n_in; (void)out_size;
  const float* x  = (const float*)d_in[0];
  const float* wb = (const float*)d_in[1];
  const float* wv = (const float*)d_in[2];
  float* out = (float*)d_out;
  const size_t per = (size_t)kBT * kN;
  float* out_marg = out;
  float* out_val  = out + per;
  float* out_cum  = out + 2 * per;

  const size_t need = (size_t)2 * kN * kF * sizeof(_Float16);  // 1 MB f16 weights
  if (d_ws != nullptr && ws_size >= need) {
    _Float16* w16 = (_Float16*)d_ws;
    convert_w_kernel<<<(kN * kF + 255) / 256, 256, 0, stream>>>(wb, wv, w16);
    fused_kernel<true><<<kBT / kRows, 256, 0, stream>>>(
        x, w16, wb, wv, out_marg, out_val, out_cum);
  } else {
    fused_kernel<false><<<kBT / kRows, 256, 0, stream>>>(
        x, nullptr, wb, wv, out_marg, out_val, out_cum);
  }
}
